// AtomPosGNN_18262200943315
// MI455X (gfx1250) — matrix operand order repaired and checked by reading, not compiled, not run on hardware
//
#include <hip/hip_runtime.h>
#include <cstdint>
#include <cstddef>

// ---------------------------------------------------------------------------
// Types (plain clang ext-vectors: POD, union-safe, map to b128 loads)
// ---------------------------------------------------------------------------
typedef __attribute__((ext_vector_type(16))) __bf16         v16bf;
typedef __attribute__((ext_vector_type(8)))  float          v8f;
typedef __attribute__((ext_vector_type(4)))  unsigned int   u32x4;
typedef __attribute__((ext_vector_type(2)))  unsigned int   u32x2;
typedef __attribute__((ext_vector_type(4)))  float          f32x4;

union FragBF {
  v16bf          v;
  u32x4          q[2];
  unsigned short s[16];
};

static constexpr int HID = 128;   // hidden width (== in_feats == out_feats)

// float -> bf16, round-to-nearest-even
__device__ __forceinline__ unsigned short f2bf(float x) {
  unsigned int u = __float_as_uint(x);
  unsigned int r = (u + 0x7FFFu + ((u >> 16) & 1u)) >> 16;
  return (unsigned short)r;
}

// numerically stable softplus: max(x,0) + log1p(exp(-|x|))
__device__ __forceinline__ float softplusf(float x) {
  return fmaxf(x, 0.f) + log1pf(expf(-fabsf(x)));
}

// ---------------------------------------------------------------------------
// Prep: weighted degree -> inv_sqrt_deg, fused with optional f32->bf16
// conversion of the adjacency. The 268 MB f32 adjacency is read exactly once,
// with NON-TEMPORAL loads so the dead streaming reads don't evict the 134 MB
// bf16 copy (which we want resident in MI455X's 192 MB L2 for all 4 layers).
// One block per row, 256 threads, float4 loads.
// ---------------------------------------------------------------------------
__global__ void __launch_bounds__(256)
prep_kernel(const float* __restrict__ adj, unsigned short* __restrict__ adjb,
            float* __restrict__ invd, int n) {
  __shared__ float red[256];
  const int row = blockIdx.x;
  const f32x4* a4 = (const f32x4*)(adj + (size_t)row * n);
  u32x2* b2 = adjb ? (u32x2*)(adjb + (size_t)row * n) : nullptr;
  const int m = n >> 2;
  float s = 0.f;
  for (int j = threadIdx.x; j < m; j += 256) {
    f32x4 v = __builtin_nontemporal_load(a4 + j);   // streaming, read-once
    s += (v.x + v.y) + (v.z + v.w);
    if (b2) {
      u32x2 pk;
      pk.x = (unsigned)f2bf(v.x) | ((unsigned)f2bf(v.y) << 16);
      pk.y = (unsigned)f2bf(v.z) | ((unsigned)f2bf(v.w) << 16);
      b2[j] = pk;
    }
  }
  red[threadIdx.x] = s;
  __syncthreads();
  for (int st = 128; st > 0; st >>= 1) {
    if ((int)threadIdx.x < st) red[threadIdx.x] += red[threadIdx.x + st];
    __syncthreads();
  }
  if (threadIdx.x == 0) {
    float deg = red[0];
    invd[row] = deg > 0.f ? rsqrtf(fmaxf(deg, 1e-12f)) : 0.f;
  }
}

// ---------------------------------------------------------------------------
// Build initial features: concat(atom_pos[N,3], atom_emb[N,125]) -> [N,128]
// ---------------------------------------------------------------------------
__global__ void __launch_bounds__(256)
concat_kernel(const float* __restrict__ pos, const float* __restrict__ emb,
              float* __restrict__ feat) {
  int idx = blockIdx.x * 256 + threadIdx.x;
  int r = idx >> 7;          // /128
  int c = idx & (HID - 1);   // %128
  feat[idx] = (c < 3) ? pos[r * 3 + c] : emb[(size_t)r * 125 + (c - 3)];
}

// ---------------------------------------------------------------------------
// Per-layer transform: h1T[c][k] = bf16( invd[k] * (feat[k,:] @ W[:,c]) )
// Stored COLUMN-major (stride n along k) so GEMM B-fragments load
// contiguously. One block per row k, 128 threads = one output column each.
// ---------------------------------------------------------------------------
__global__ void __launch_bounds__(128)
transform_kernel(const float* __restrict__ feat, const float* __restrict__ W,
                 const float* __restrict__ invd, unsigned short* __restrict__ h1T,
                 int n) {
  __shared__ float frow[HID];
  const int k = blockIdx.x;
  const int c = threadIdx.x;
  frow[c] = feat[(size_t)k * HID + c] * invd[k];
  __syncthreads();
  float acc = 0.f;
#pragma unroll 8
  for (int i = 0; i < HID; ++i) acc += frow[i] * W[i * HID + c];
  h1T[(size_t)c * n + k] = f2bf(acc);
}

// ---------------------------------------------------------------------------
// Fragment loaders, matching the CDNA5 16-bit WMMA VGPR layouts.
// A (16x32, MxK, row-major mem): lane half 0 -> K {kb..kb+7, kb+16..kb+23},
//                                lane half 1 -> K {kb+8..kb+15, kb+24..kb+31}
// ---------------------------------------------------------------------------
__device__ __forceinline__ void load_a_frag(FragBF& f, const unsigned short* rowp,
                                            int kb, int half) {
  f.q[0] = *(const u32x4*)(rowp + kb + half * 8);
  f.q[1] = *(const u32x4*)(rowp + kb + 16 + half * 8);
}

// f32 adjacency fallback: stream (non-temporal) 16 floats, convert in-register.
// 268 MB/layer can never be L2-resident, so keep it out of L2 to protect the
// hot h1T working set.
__device__ __forceinline__ void load_a_frag(FragBF& f, const float* rowp,
                                            int kb, int half) {
  const f32x4* p0 = (const f32x4*)(rowp + kb + half * 8);
  const f32x4* p1 = (const f32x4*)(rowp + kb + 16 + half * 8);
  f32x4 a = __builtin_nontemporal_load(p0);
  f32x4 b = __builtin_nontemporal_load(p0 + 1);
  f32x4 c = __builtin_nontemporal_load(p1);
  f32x4 d = __builtin_nontemporal_load(p1 + 1);
  f.s[0]  = f2bf(a.x); f.s[1]  = f2bf(a.y); f.s[2]  = f2bf(a.z); f.s[3]  = f2bf(a.w);
  f.s[4]  = f2bf(b.x); f.s[5]  = f2bf(b.y); f.s[6]  = f2bf(b.z); f.s[7]  = f2bf(b.w);
  f.s[8]  = f2bf(c.x); f.s[9]  = f2bf(c.y); f.s[10] = f2bf(c.z); f.s[11] = f2bf(c.w);
  f.s[12] = f2bf(d.x); f.s[13] = f2bf(d.y); f.s[14] = f2bf(d.z); f.s[15] = f2bf(d.w);
}

// B (32x16, KxN): lanes 0-15 = col N, K {kb..kb+7, kb+16..kb+23}; lanes 16-31 = K {kb+8..kb+15, kb+24..kb+31}.
// h1T is column-major so this is two 16-byte chunks per lane.
__device__ __forceinline__ void load_b_frag(FragBF& f, const unsigned short* colp,
                                            int kb, int half) {
  f.q[0] = *(const u32x4*)(colp + kb + half * 8);
  f.q[1] = *(const u32x4*)(colp + kb + 16 + half * 8);
}

// ---------------------------------------------------------------------------
// Aggregation GEMM + fused epilogue:
//   out[r,c] = softplus( invd[r] * sum_k adj[r,k]*h1[k,c] + bias[c] )
// Wave: 2x2 tiles of 16x16 (rows 32, cols 32). Block: 8 waves =
// 2 (M) x 4 (N) -> 64 rows x 128 cols. Grid: n/64 blocks.
// ---------------------------------------------------------------------------
template <typename AT>
__global__ void __launch_bounds__(256)
gemm_layer_kernel(const AT* __restrict__ adjA, const unsigned short* __restrict__ h1T,
                  const float* __restrict__ invd, const float* __restrict__ bias,
                  float* __restrict__ outF, int n) {
  const int lane  = threadIdx.x & 31;
  const int wave  = threadIdx.x >> 5;
  const int waveM = wave & 1;   // 2 waves along M
  const int waveN = wave >> 1;  // 4 waves along N
  const int half  = lane >> 4;
  const int l15   = lane & 15;

  const int rowBase = blockIdx.x * 64 + waveM * 32;  // wave owns rows [rowBase, rowBase+32)
  const int colBase = waveN * 32;                    // wave owns cols [colBase, colBase+32)

  const AT* aptr0 = adjA + (size_t)(rowBase + l15) * n;        // M-tile 0 rows
  const AT* aptr1 = adjA + (size_t)(rowBase + 16 + l15) * n;   // M-tile 1 rows
  const unsigned short* bptr0 = h1T + (size_t)(colBase + l15) * n;       // N-tile 0 cols
  const unsigned short* bptr1 = h1T + (size_t)(colBase + 16 + l15) * n;  // N-tile 1 cols

  v8f acc00 = {}, acc01 = {}, acc10 = {}, acc11 = {};

  for (int kb = 0; kb < n; kb += 32) {
    FragBF A0, A1, B0, B1;
    load_a_frag(A0, aptr0, kb, half);
    load_a_frag(A1, aptr1, kb, half);
    load_b_frag(B0, bptr0, kb, half);
    load_b_frag(B1, bptr1, kb, half);
    acc00 = __builtin_amdgcn_wmma_f32_16x16x32_bf16(false, A0.v, false, B0.v,
                                                    (short)0, acc00, false, false);
    acc01 = __builtin_amdgcn_wmma_f32_16x16x32_bf16(false, A0.v, false, B1.v,
                                                    (short)0, acc01, false, false);
    acc10 = __builtin_amdgcn_wmma_f32_16x16x32_bf16(false, A1.v, false, B0.v,
                                                    (short)0, acc10, false, false);
    acc11 = __builtin_amdgcn_wmma_f32_16x16x32_bf16(false, A1.v, false, B1.v,
                                                    (short)0, acc11, false, false);
  }

  // Epilogue. C/D layout: VGPR r, lanes 0-15 -> M = r, lanes 16-31 -> M = r+8,
  // N = lane%16.
  const float bias0 = bias[colBase + l15];
  const float bias1 = bias[colBase + 16 + l15];
  const int rbase0 = rowBase + half * 8;        // M-tile 0
  const int rbase1 = rowBase + 16 + half * 8;   // M-tile 1
#pragma unroll
  for (int r = 0; r < 8; ++r) {
    const int r0 = rbase0 + r, r1 = rbase1 + r;
    const float d0 = invd[r0], d1 = invd[r1];
    outF[(size_t)r0 * HID + colBase + l15]      = softplusf(acc00[r] * d0 + bias0);
    outF[(size_t)r0 * HID + colBase + 16 + l15] = softplusf(acc01[r] * d0 + bias1);
    outF[(size_t)r1 * HID + colBase + l15]      = softplusf(acc10[r] * d1 + bias0);
    outF[(size_t)r1 * HID + colBase + 16 + l15] = softplusf(acc11[r] * d1 + bias1);
  }
}

// ---------------------------------------------------------------------------
// Host launcher
// ---------------------------------------------------------------------------
extern "C" void kernel_launch(void* const* d_in, const int* in_sizes, int n_in,
                              void* d_out, int out_size, void* d_ws, size_t ws_size,
                              hipStream_t stream) {
  const float* atom_pos = (const float*)d_in[0];  // [N,3]
  const float* atom_emb = (const float*)d_in[1];  // [N,125]
  const float* dist_adj = (const float*)d_in[2];  // [N,N]
  const float* Ws       = (const float*)d_in[3];  // [L,128,128]
  const float* bs       = (const float*)d_in[4];  // [L,128]

  const int N = in_sizes[0] / 3;                  // 8192
  const int L = in_sizes[3] / (HID * HID);        // 4

  char* ws = (char*)d_ws;
  size_t off = 0;
  auto take = [&](size_t bytes) -> char* {
    char* p = ws + off;
    off += (bytes + 255) & ~(size_t)255;
    return p;
  };

  float*          invd  = (float*)take((size_t)N * sizeof(float));
  unsigned short* h1T   = (unsigned short*)take((size_t)HID * N * sizeof(unsigned short));
  float*          feat0 = (float*)take((size_t)N * HID * sizeof(float));
  float*          feat1 = (float*)take((size_t)N * HID * sizeof(float));

  // bf16 adjacency copy (134 MB) -> fits MI455X's 192 MB L2; use if ws allows.
  const size_t adjBytes = (size_t)N * N * sizeof(unsigned short);
  unsigned short* adjb = nullptr;
  if (off + adjBytes <= ws_size) adjb = (unsigned short*)take(adjBytes);

  // 1) degree/rsqrt (+ one-shot f32->bf16 adjacency conversion)
  prep_kernel<<<dim3(N), dim3(256), 0, stream>>>(dist_adj, adjb, invd, N);
  // 2) initial features
  concat_kernel<<<dim3((N * HID) / 256), dim3(256), 0, stream>>>(atom_pos, atom_emb, feat0);

  float* cur = feat0;
  float* nxt = feat1;
  for (int l = 0; l < L; ++l) {
    transform_kernel<<<dim3(N), dim3(HID), 0, stream>>>(
        cur, Ws + (size_t)l * HID * HID, invd, h1T, N);
    float* outp = (l == L - 1) ? (float*)d_out : nxt;
    const dim3 grid(N / 64), block(256);
    if (adjb) {
      gemm_layer_kernel<unsigned short><<<grid, block, 0, stream>>>(
          adjb, h1T, invd, bs + (size_t)l * HID, outp, N);
    } else {
      gemm_layer_kernel<float><<<grid, block, 0, stream>>>(
          dist_adj, h1T, invd, bs + (size_t)l * HID, outp, N);
    }
    float* t = cur; cur = nxt; nxt = t;
  }
}
